// ModifiedLSTM_60997125538430
// MI455X (gfx1250) — matrix operand order repaired and checked by reading, not compiled, not run on hardware
//
#include <hip/hip_runtime.h>

typedef __attribute__((ext_vector_type(16))) __bf16 v16bf;
typedef __attribute__((ext_vector_type(8)))  __bf16 v8bf;
typedef __attribute__((ext_vector_type(8)))  float  v8f;
typedef __attribute__((ext_vector_type(4)))  float  v4f;

union BF16x16 { v16bf v; v8bf h[2]; };
union BF8     { v8bf v; __bf16 e[8]; };

#define BB   64
#define TT   512
#define DD   256
#define HH   512
#define FH   2048   // 4*H
#define BT   (BB*TT)

// CDNA5 has a hardware transcendental V_TANH_F32 — use it when the toolchain
// exposes the builtin; otherwise a branch-free __expf-based fallback.
#if __has_builtin(__builtin_amdgcn_tanhf)
__device__ __forceinline__ float fast_tanh(float x) { return __builtin_amdgcn_tanhf(x); }
#else
__device__ __forceinline__ float fast_tanh(float x)
{
    return 2.0f / (1.0f + __expf(-2.0f * x)) - 1.0f;
}
#endif
__device__ __forceinline__ float sigmoidf_(float x) { return 0.5f * fast_tanh(0.5f * x) + 0.5f; }

// Fragment loaders. Lane base pointers already include the +16*lh (A) / +16*lh (B)
// per-lane component, so every load is base + compile-time immediate offset.
__device__ __forceinline__ void ldA(BF16x16& f, const __bf16* p)
{
    f.h[0] = *(const v8bf*)(p);
    f.h[1] = *(const v8bf*)(p + 8);
}
__device__ __forceinline__ void ldB(BF16x16& f, const __bf16* p)
{
    f.h[0] = *(const v8bf*)(p);
    f.h[1] = *(const v8bf*)(p + 8);
}
__device__ __forceinline__ v8f wmma_bf16(const BF16x16& a, const BF16x16& b, v8f c)
{
    return __builtin_amdgcn_wmma_f32_16x16x32_bf16(false, a.v, false, b.v,
                                                   (short)0, c, false, false);
}

// ---------------------------------------------------------------------------
// Generic GEMM: out[M x N] = A[M x K](bf16) * W[N x K]^T(bf16) + bias[N]
// One wave computes a 16(M) x 64(N) tile; A fragment reused across 4 N-subtiles.
// K is compile time: full unroll, immediate offsets, software-pipelined.
// out_bf != null -> bf16 output with ld = N; else fp32 output, cols < nvalid.
// ---------------------------------------------------------------------------
template<int K>
__global__ void wmma_gemm_bias(const __bf16* __restrict__ A,
                               const __bf16* __restrict__ W,
                               const float* __restrict__ bias,
                               int M, int N,
                               __bf16* __restrict__ out_bf,
                               float* __restrict__ out_f,
                               int nvalid)
{
    const int lane = threadIdx.x & 31;
    const int wid  = (blockIdx.x * blockDim.x + threadIdx.x) >> 5;
    const int nblk64 = N >> 6;
    const int mtile  = wid / nblk64;
    if (mtile >= (M >> 4)) return;                 // wave-uniform early out
    const int nblk = wid - mtile * nblk64;
    const int M0 = mtile << 4;
    const int N0 = nblk  << 6;
    const int lm = lane & 15, lh = lane >> 4;

    const __bf16* Ap = A + (size_t)(M0 + lm) * K + 16 * lh;
    const __bf16* Bp[4];
    #pragma unroll
    for (int s = 0; s < 4; ++s)
        Bp[s] = W + (size_t)(N0 + s * 16 + lm) * K + 16 * lh;

    v8f acc[4] = {};
    BF16x16 a0, a1, b0[4], b1[4];

    ldA(a0, Ap);
    #pragma unroll
    for (int s = 0; s < 4; ++s) ldB(b0[s], Bp[s]);

    #pragma unroll
    for (int k0 = 0; k0 < K; k0 += 64) {
        ldA(a1, Ap + k0 + 32);
        #pragma unroll
        for (int s = 0; s < 4; ++s) ldB(b1[s], Bp[s] + k0 + 32);
        if (k0 + 128 <= K) __builtin_prefetch(Ap + k0 + 128, 0, 3);
        #pragma unroll
        for (int s = 0; s < 4; ++s) acc[s] = wmma_bf16(a0, b0[s], acc[s]);

        if (k0 + 64 < K) {
            ldA(a0, Ap + k0 + 64);
            #pragma unroll
            for (int s = 0; s < 4; ++s) ldB(b0[s], Bp[s] + k0 + 64);
        }
        #pragma unroll
        for (int s = 0; s < 4; ++s) acc[s] = wmma_bf16(a1, b1[s], acc[s]);
    }

    #pragma unroll
    for (int s = 0; s < 4; ++s) {
        const int col = N0 + s * 16 + lm;
        const float bv = bias[col];
        #pragma unroll
        for (int j = 0; j < 8; ++j) {
            const int m = M0 + j + 8 * lh;
            const float v = acc[s][j] + bv;
            if (out_bf) {
                out_bf[(size_t)m * N + col] = (__bf16)v;
            } else if (col < nvalid) {
                out_f[(size_t)m * nvalid + col] = v;
            }
        }
    }
}

// ---------------------------------------------------------------------------
// Fused recurrent step: gates = h_in @ Whh^T + G[:,t,:]; cell update; mask.
// One BLOCK (512 threads = 16 waves) owns a 16(batch) x 16(hcol) tile.
// Wave wv: gate = wv&3, K-quarter = wv>>2 -> each wave does a K=128 WMMA
// chain (4 blocks deep). Partials are combined through LDS, wave 0 applies
// the sigmoid/tanh cell update. 2048 waves per step -> real latency hiding.
// h double-buffered across steps (h_in read by all, h_out by owner lane).
// ---------------------------------------------------------------------------
__global__ void lstm_step(const __bf16* __restrict__ Whh,  // [4H x H]
                          const __bf16* __restrict__ G,    // [B*T x 4H] x-part + bias
                          const float* __restrict__ mask,  // [B x T]
                          const __bf16* __restrict__ h_in, // [B x H]
                          __bf16* __restrict__ h_out,      // [B x H]
                          float*  __restrict__ c,          // [B x H]
                          __bf16* __restrict__ hseq,       // [B x T x H] or null
                          float*  __restrict__ hlast,      // [B x H] or null
                          int t)
{
    __shared__ float red[16][8][32];   // [wave][j][lane], lane-major: conflict-free
    __shared__ float gx[4][8][32];     // x-part gate pre-activations

    const int lane = threadIdx.x & 31;
    const int wv   = threadIdx.x >> 5;        // 0..15
    const int gate = wv & 3;
    const int kq   = wv >> 2;                 // 0..3
    const int mtile = blockIdx.x >> 5;        // 0..3   (B/16)
    const int ntile = blockIdx.x & 31;        // 0..31  (H/16)
    const int M0 = mtile << 4;
    const int N0 = ntile << 4;
    const int lm = lane & 15, lh = lane >> 4;
    const int Kb = kq << 7;                   // kq*128

    // kq==0 waves pre-stage G[:,t,:] for their gate into LDS; these loads
    // overlap the WMMA phase of all 16 waves.
    if (kq == 0) {
        #pragma unroll
        for (int j = 0; j < 8; ++j) {
            const int m = M0 + j + 8 * lh;
            gx[gate][j][lane] =
                (float)G[((size_t)m * TT + t) * FH + gate * HH + N0 + lm];
        }
    }

    const __bf16* Ap = h_in + (size_t)(M0 + lm) * HH + Kb + 16 * lh;
    const __bf16* Bp = Whh + (size_t)(gate * HH + N0 + lm) * HH + Kb + 16 * lh;

    v8f acc = {};
    BF16x16 a0, a1, bb0, bb1;
    ldA(a0, Ap);       ldB(bb0, Bp);
    ldA(a1, Ap + 32);  ldB(bb1, Bp + 32);
    acc = wmma_bf16(a0, bb0, acc);
    ldA(a0, Ap + 64);  ldB(bb0, Bp + 64);
    acc = wmma_bf16(a1, bb1, acc);
    ldA(a1, Ap + 96);  ldB(bb1, Bp + 96);
    acc = wmma_bf16(a0, bb0, acc);
    acc = wmma_bf16(a1, bb1, acc);

    #pragma unroll
    for (int j = 0; j < 8; ++j) red[wv][j][lane] = acc[j];
    __syncthreads();

    if (wv == 0) {
        const int ncol = N0 + lm;
        #pragma unroll
        for (int j = 0; j < 8; ++j) {
            const int m = M0 + j + 8 * lh;
            float gv4[4];
            #pragma unroll
            for (int g = 0; g < 4; ++g) {
                gv4[g] = red[g][j][lane] + red[4 + g][j][lane]
                       + red[8 + g][j][lane] + red[12 + g][j][lane]
                       + gx[g][j][lane];
            }
            const float iv = sigmoidf_(gv4[0]);
            const float fv = sigmoidf_(gv4[1]);
            const float gg = fast_tanh(gv4[2]);
            const float ov = sigmoidf_(gv4[3]);

            const size_t hidx = (size_t)m * HH + ncol;
            float cv = fv * c[hidx] + iv * gg;
            float hv = ov * fast_tanh(cv);
            const float mt = mask[(size_t)m * TT + t];
            hv *= mt;
            cv *= mt;
            c[hidx] = cv;
            h_out[hidx] = (__bf16)hv;
            if (hseq)  hseq[((size_t)m * TT + t) * HH + ncol] = (__bf16)hv;
            if (hlast) hlast[hidx] = hv;
        }
    }
}

// ---------------------------------------------------------------------------
// ReLU + LayerNorm over H=512. One wave per row; each lane owns a contiguous
// 16-element chunk (vectorized b128 loads/stores; LN sums are order-invariant).
// ---------------------------------------------------------------------------
__global__ void relu_ln_bf(const __bf16* __restrict__ in, const float* __restrict__ g,
                           const float* __restrict__ be, __bf16* __restrict__ out, int R)
{
    const int wid  = (blockIdx.x * blockDim.x + threadIdx.x) >> 5;
    const int lane = threadIdx.x & 31;
    if (wid >= R) return;
    const __bf16* row = in + (size_t)wid * HH + lane * 16;
    BF8 r0, r1;
    r0.v = *(const v8bf*)(row);
    r1.v = *(const v8bf*)(row + 8);
    float v[16], s = 0.f, ss = 0.f;
    #pragma unroll
    for (int i = 0; i < 8; ++i) {
        v[i]     = fmaxf((float)r0.e[i], 0.f);
        v[i + 8] = fmaxf((float)r1.e[i], 0.f);
    }
    #pragma unroll
    for (int i = 0; i < 16; ++i) { s += v[i]; ss += v[i] * v[i]; }
    #pragma unroll
    for (int off = 16; off > 0; off >>= 1) {
        s  += __shfl_xor(s,  off, 32);
        ss += __shfl_xor(ss, off, 32);
    }
    const float mu  = s * (1.f / HH);
    const float var = ss * (1.f / HH) - mu * mu;
    const float rstd = rsqrtf(var + 1e-5f);
    v4f gv[4], bv[4];
    #pragma unroll
    for (int q = 0; q < 4; ++q) {
        gv[q] = *(const v4f*)(g  + lane * 16 + 4 * q);
        bv[q] = *(const v4f*)(be + lane * 16 + 4 * q);
    }
    BF8 o0, o1;
    #pragma unroll
    for (int i = 0; i < 8; ++i) {
        o0.e[i] = (__bf16)((v[i]     - mu) * rstd * gv[i >> 2][i & 3] + bv[i >> 2][i & 3]);
        o1.e[i] = (__bf16)((v[i + 8] - mu) * rstd * gv[2 + (i >> 2)][i & 3] + bv[2 + (i >> 2)][i & 3]);
    }
    __bf16* orow = out + (size_t)wid * HH + lane * 16;
    *(v8bf*)(orow)     = o0.v;
    *(v8bf*)(orow + 8) = o1.v;
}

__global__ void relu_ln_f32(const float* __restrict__ in, const float* __restrict__ g,
                            const float* __restrict__ be, __bf16* __restrict__ out, int R)
{
    const int wid  = (blockIdx.x * blockDim.x + threadIdx.x) >> 5;
    const int lane = threadIdx.x & 31;
    if (wid >= R) return;
    const float* row = in + (size_t)wid * HH + lane * 16;
    v4f r[4];
    #pragma unroll
    for (int q = 0; q < 4; ++q) r[q] = *(const v4f*)(row + 4 * q);
    float v[16], s = 0.f, ss = 0.f;
    #pragma unroll
    for (int i = 0; i < 16; ++i) {
        v[i] = fmaxf(r[i >> 2][i & 3], 0.f);
        s += v[i]; ss += v[i] * v[i];
    }
    #pragma unroll
    for (int off = 16; off > 0; off >>= 1) {
        s  += __shfl_xor(s,  off, 32);
        ss += __shfl_xor(ss, off, 32);
    }
    const float mu  = s * (1.f / HH);
    const float var = ss * (1.f / HH) - mu * mu;
    const float rstd = rsqrtf(var + 1e-5f);
    v4f gv[4], bv[4];
    #pragma unroll
    for (int q = 0; q < 4; ++q) {
        gv[q] = *(const v4f*)(g  + lane * 16 + 4 * q);
        bv[q] = *(const v4f*)(be + lane * 16 + 4 * q);
    }
    BF8 o0, o1;
    #pragma unroll
    for (int i = 0; i < 8; ++i) {
        o0.e[i] = (__bf16)((v[i]     - mu) * rstd * gv[i >> 2][i & 3] + bv[i >> 2][i & 3]);
        o1.e[i] = (__bf16)((v[i + 8] - mu) * rstd * gv[2 + (i >> 2)][i & 3] + bv[2 + (i >> 2)][i & 3]);
    }
    __bf16* orow = out + (size_t)wid * HH + lane * 16;
    *(v8bf*)(orow)     = o0.v;
    *(v8bf*)(orow + 8) = o1.v;
}

// ---------------------------------------------------------------------------
// Small utilities
// ---------------------------------------------------------------------------
__global__ void f2bf(const float* __restrict__ in, __bf16* __restrict__ out, int n)
{
    int i = blockIdx.x * blockDim.x + threadIdx.x;
    if (i < n) out[i] = (__bf16)in[i];
}

__global__ void bias_sum(const float* __restrict__ a, const float* __restrict__ b,
                         float* __restrict__ out, int n)
{
    int i = blockIdx.x * blockDim.x + threadIdx.x;
    if (i < n) out[i] = a[i] + b[i];
}

__global__ void pad_fc(const float* __restrict__ fcW, const float* __restrict__ fcb,
                       __bf16* __restrict__ Wb, float* __restrict__ bb)
{
    int i = blockIdx.x * blockDim.x + threadIdx.x;  // covers 1024*512
    if (i < 1024 * 512) {
        int r = i >> 9;
        Wb[i] = (r < 1000) ? (__bf16)fcW[i] : (__bf16)0.0f;
    }
    if (i < 1024) bb[i] = (i < 1000) ? fcb[i] : 0.0f;
}

__global__ void zero_bf(__bf16* p, int n)
{
    int i = blockIdx.x * blockDim.x + threadIdx.x;
    if (i < n) p[i] = (__bf16)0.0f;
}

__global__ void zero_f(float* p, int n)
{
    int i = blockIdx.x * blockDim.x + threadIdx.x;
    if (i < n) p[i] = 0.0f;
}

// ---------------------------------------------------------------------------
extern "C" void kernel_launch(void* const* d_in, const int* in_sizes, int n_in,
                              void* d_out, int out_size, void* d_ws, size_t ws_size,
                              hipStream_t stream)
{
    (void)in_sizes; (void)n_in; (void)out_size; (void)ws_size;
    const float* x    = (const float*)d_in[0];
    const float* rm   = (const float*)d_in[1];
    const float* Wih0 = (const float*)d_in[2];
    const float* Whh0 = (const float*)d_in[3];
    const float* bih0 = (const float*)d_in[4];
    const float* bhh0 = (const float*)d_in[5];
    const float* g0   = (const float*)d_in[6];
    const float* be0  = (const float*)d_in[7];
    const float* Wih1 = (const float*)d_in[8];
    const float* Whh1 = (const float*)d_in[9];
    const float* bih1 = (const float*)d_in[10];
    const float* bhh1 = (const float*)d_in[11];
    const float* g1   = (const float*)d_in[12];
    const float* be1  = (const float*)d_in[13];
    const float* fcW  = (const float*)d_in[14];
    const float* fcb  = (const float*)d_in[15];
    float* out = (float*)d_out;

    char* p = (char*)d_ws;
    auto carve = [&](size_t bytes) -> char* {
        char* r = p;
        p += (bytes + 255) & ~(size_t)255;
        return r;
    };

    __bf16* xb    = (__bf16*)carve((size_t)BT * DD * 2);
    __bf16* Wih0b = (__bf16*)carve((size_t)FH * DD * 2);
    __bf16* Whh0b = (__bf16*)carve((size_t)FH * HH * 2);
    __bf16* Wih1b = (__bf16*)carve((size_t)FH * HH * 2);
    __bf16* Whh1b = (__bf16*)carve((size_t)FH * HH * 2);
    __bf16* fcWb  = (__bf16*)carve((size_t)1024 * HH * 2);
    float*  bias0 = (float*)carve((size_t)FH * 4);
    float*  bias1 = (float*)carve((size_t)FH * 4);
    float*  fcbp  = (float*)carve((size_t)1024 * 4);
    __bf16* Gbuf  = (__bf16*)carve((size_t)BT * FH * 2);   // 128 MB, reused per layer
    __bf16* h0seq = (__bf16*)carve((size_t)BT * HH * 2);
    __bf16* A1    = (__bf16*)carve((size_t)BT * HH * 2);
    __bf16* hA    = (__bf16*)carve((size_t)BB * HH * 2);
    __bf16* hB    = (__bf16*)carve((size_t)BB * HH * 2);
    float*  cbuf  = (float*)carve((size_t)BB * HH * 4);
    float*  h1l   = (float*)carve((size_t)BB * HH * 4);
    __bf16* A2    = (__bf16*)carve((size_t)BB * HH * 2);

    auto cdiv = [](int a, int b) { return (a + b - 1) / b; };

    // fp32 -> bf16 conversions (weights + x), bias sums, padded classifier weights
    f2bf<<<cdiv(BT * DD, 256), 256, 0, stream>>>(x, xb, BT * DD);
    f2bf<<<cdiv(FH * DD, 256), 256, 0, stream>>>(Wih0, Wih0b, FH * DD);
    f2bf<<<cdiv(FH * HH, 256), 256, 0, stream>>>(Whh0, Whh0b, FH * HH);
    f2bf<<<cdiv(FH * HH, 256), 256, 0, stream>>>(Wih1, Wih1b, FH * HH);
    f2bf<<<cdiv(FH * HH, 256), 256, 0, stream>>>(Whh1, Whh1b, FH * HH);
    bias_sum<<<cdiv(FH, 256), 256, 0, stream>>>(bih0, bhh0, bias0, FH);
    bias_sum<<<cdiv(FH, 256), 256, 0, stream>>>(bih1, bhh1, bias1, FH);
    pad_fc<<<cdiv(1024 * 512, 256), 256, 0, stream>>>(fcW, fcb, fcWb, fcbp);

    // ---- layer 0: time-parallel x-gates GEMM, then 512 fused recurrent steps
    {
        int waves = (FH / 64) * (BT / 16);
        wmma_gemm_bias<DD><<<cdiv(waves * 32, 256), 256, 0, stream>>>(
            xb, Wih0b, bias0, BT, FH, Gbuf, nullptr, 0);
    }
    zero_bf<<<cdiv(BB * HH, 256), 256, 0, stream>>>(hA, BB * HH);
    zero_f<<<cdiv(BB * HH, 256), 256, 0, stream>>>(cbuf, BB * HH);
    for (int t = 0; t < TT; ++t) {
        const __bf16* hin = (t & 1) ? hB : hA;
        __bf16*      hout = (t & 1) ? hA : hB;
        lstm_step<<<128, 512, 0, stream>>>(Whh0b, Gbuf, rm, hin, hout, cbuf, h0seq, nullptr, t);
    }
    relu_ln_bf<<<cdiv(BT * 32, 256), 256, 0, stream>>>(h0seq, g0, be0, A1, BT);

    // ---- layer 1
    {
        int waves = (FH / 64) * (BT / 16);
        wmma_gemm_bias<HH><<<cdiv(waves * 32, 256), 256, 0, stream>>>(
            A1, Wih1b, bias1, BT, FH, Gbuf, nullptr, 0);
    }
    zero_bf<<<cdiv(BB * HH, 256), 256, 0, stream>>>(hA, BB * HH);
    zero_f<<<cdiv(BB * HH, 256), 256, 0, stream>>>(cbuf, BB * HH);
    for (int t = 0; t < TT; ++t) {
        const __bf16* hin = (t & 1) ? hB : hA;
        __bf16*      hout = (t & 1) ? hA : hB;
        lstm_step<<<128, 512, 0, stream>>>(Whh1b, Gbuf, rm, hin, hout, cbuf, nullptr, h1l, t);
    }
    relu_ln_f32<<<cdiv(BB * 32, 256), 256, 0, stream>>>(h1l, g1, be1, A2, BB);

    // ---- classifier: [64 x 512] @ [1024(pad) x 512]^T -> fp32 out, cols < 1000
    {
        int waves = (1024 / 64) * (BB / 16);
        wmma_gemm_bias<HH><<<cdiv(waves * 32, 256), 256, 0, stream>>>(
            A2, fcWb, fcbp, BB, 1024, nullptr, out, 1000);
    }
}
